// localAttn_29523605192891
// MI455X (gfx1250) — matrix operand order repaired and checked by reading, not compiled, not run on hardware
//
#include <hip/hip_runtime.h>

// ---------------------------------------------------------------------------
// Windowed local attention (Swin-style) for MI455X / gfx1250, wave32 + WMMA.
//   x:(4,256,256,256) f32 -> 4096 windows of (64 tokens x 256 ch)
//   per window: qkv proj, 8-head attention (seq=64, dhead=32), out proj.
// All matmuls on v_wmma_f32_16x16x32_bf16 (bf16 in, f32 accum).
// Round 2: capped unrolling + per-strip softmax to keep VGPRs < 256
// (round 1 spilled into the s_set_vgpr_msb window -> 1 wave/SIMD).
// ---------------------------------------------------------------------------

typedef __attribute__((ext_vector_type(16))) __bf16       v16bf;
typedef __attribute__((ext_vector_type(8)))  float        v8f;
typedef __attribute__((ext_vector_type(4)))  unsigned int u32x4;

#define WMMA_BF16(A, B, C) \
  __builtin_amdgcn_wmma_f32_16x16x32_bf16(false, (A), false, (B), (short)0, (C), false, false)

__device__ __forceinline__ unsigned short f2bf(float f) {
  unsigned u = __float_as_uint(f);
  u += 0x7FFFu + ((u >> 16) & 1u);          // round-to-nearest-even
  return (unsigned short)(u >> 16);
}

// A-matrix tile load (16x32 bf16): lane holds row M=lane%16.
// Lanes 0-15: K = 0..15, lanes 16-31: K = 16..31 (contiguous 32B).
// p must point at (row base + kbase + half*16); 16B aligned.
__device__ __forceinline__ v16bf load_a(const unsigned short* p) {
  union { v16bf v; u32x4 u[2]; } t;
  t.u[0] = *(const u32x4*)(p);
  t.u[1] = *(const u32x4*)(p + 8);
  return t.v;
}

// B-matrix tile load (32x16 bf16): lane holds col N=lane%16.
// Lanes 0-15: K = 0..15, lanes 16-31: K = 16..31 (contiguous 32B).
// p must point at (col-row base + kbase + half*16); 16B aligned.
__device__ __forceinline__ v16bf load_b(const unsigned short* p) {
  union { v16bf v; u32x4 u[2]; } t;
  t.u[0] = *(const u32x4*)(p);
  t.u[1] = *(const u32x4*)(p + 8);
  return t.v;
}

// ---------------------------------------------------------------------------
// Pre-pass: fp32 weights -> bf16, transposed to N-major so WMMA B-tiles are
// contiguous 32B loads. ws layout (ushort): WqT[256*256] | WkvT[512*256] |
// WpT[256*256]  (512 KB total, L2-resident, reused by all 4096 blocks).
// ---------------------------------------------------------------------------
__global__ void cvt_weights(const float* __restrict__ Wq,
                            const float* __restrict__ Wkv,
                            const float* __restrict__ Wp,
                            unsigned short* __restrict__ wsT) {
  int t = blockIdx.x * blockDim.x + threadIdx.x;   // 0 .. 262143
  if (t < 65536) {
    int n = t >> 8, k = t & 255;
    wsT[t] = f2bf(Wq[k * 256 + n]);
  } else if (t < 196608) {
    int i = t - 65536;
    int n = i >> 8, k = i & 255;                   // n in [0,512)
    wsT[t] = f2bf(Wkv[k * 512 + n]);
  } else {
    int i = t - 196608;
    int n = i >> 8, k = i & 255;
    wsT[t] = f2bf(Wp[k * 256 + n]);
  }
}

// ---------------------------------------------------------------------------
// Main kernel: one block per window, 8 waves (one head per wave).
// Dynamic LDS layout (ushort elements, all regions 16B aligned):
//   xw  : 64 x 264      window activations, bf16 (8-elem row pad)
//   qh  : 8 x 64 x 40   q per head, pre-scaled, row-major (seq, d)
//   kh  : 8 x 64 x 40   k per head, row-major (seq, d)
//   vt  : 8 x 32 x 72   v per head, TRANSPOSED (d, seq) -> direct B-matrix
//   att : 8 x 64 x 72   softmax probs bf16 (re-layout bounce for A-matrix)
//   oat : 64 x 264      attention output, bf16 A-matrix for final proj
// Total ~254 KB of the 320 KB/WG budget.
// ---------------------------------------------------------------------------
__global__ __launch_bounds__(256, 2) void win_attn(
    const float* __restrict__ x, const float* __restrict__ pos,
    const float* __restrict__ bp,
    const unsigned short* __restrict__ WqT,
    const unsigned short* __restrict__ WkvT,
    const unsigned short* __restrict__ WpT,
    float* __restrict__ out) {
  extern __shared__ unsigned short smem[];
  unsigned short* xw  = smem;                 // 16896
  unsigned short* qh  = xw  + 64 * 264;       // 20480
  unsigned short* kh  = qh  + 8 * 64 * 40;    // 20480
  unsigned short* vt  = kh  + 8 * 64 * 40;    // 18432
  unsigned short* att = vt  + 8 * 32 * 72;    // 36864
  unsigned short* oat = att + 8 * 64 * 72;    // 16896  (total 130048 halves)

  const int tid  = threadIdx.x;
  const int lane = tid & 31;
  const int wid  = tid >> 5;                  // wave id == head id
  const int half = lane >> 4;
  const int lm   = lane & 15;

  const int win = blockIdx.x;                 // 0..4095
  const int bi  = win >> 10;
  const int nh  = (win >> 5) & 31;
  const int nw  = win & 31;

  // ---- Stage 1: window activations -> LDS bf16 (HBM read of x, once) ----
#pragma unroll
  for (int it = 0; it < 16; ++it) {
    int flat = it * 256 + tid;                // float4 id in window, 0..4095
    int row  = flat >> 6;                     // seq index 0..63
    int c4   = (flat & 63) << 2;              // channel 0..252
    int w0 = row >> 3, w1 = row & 7;
    const float4 v = *(const float4*)(
        x + ((((long)bi * 256 + nh * 8 + w0) * 256) + nw * 8 + w1) * 256 + c4);
    unsigned short* d = xw + row * 264 + c4;
    unsigned p0 = (unsigned)f2bf(v.x) | ((unsigned)f2bf(v.y) << 16);
    unsigned p1 = (unsigned)f2bf(v.z) | ((unsigned)f2bf(v.w) << 16);
    *(unsigned*)(d)     = p0;
    *(unsigned*)(d + 2) = p1;
  }
  __syncthreads();

  const int h = wid;
  const float qscale = 0.17677669529663687f;  // dhead^-0.5

  // ---- Stage 2: q/k/v projection for this wave's head ----
  // One tile-triple live at a time (3 accumulators); A-tile shared x3 WMMAs.
#pragma unroll 1
  for (int mt = 0; mt < 4; ++mt) {
#pragma unroll 1
    for (int nt = 0; nt < 2; ++nt) {
      v8f aq = {0,0,0,0,0,0,0,0};
      v8f ak = {0,0,0,0,0,0,0,0};
      v8f av = {0,0,0,0,0,0,0,0};
      const int nrow = h * 32 + nt * 16 + lm; // output channel (B-matrix col)
#pragma unroll
      for (int kb = 0; kb < 8; ++kb) {
        v16bf a  = load_a(xw + (mt * 16 + lm) * 264 + kb * 32 + half * 16);
        v16bf bq = load_b(WqT  + nrow * 256         + kb * 32 + half * 16);
        v16bf bk = load_b(WkvT + nrow * 256         + kb * 32 + half * 16);
        v16bf bv = load_b(WkvT + (256 + nrow) * 256 + kb * 32 + half * 16);
        aq = WMMA_BF16(a, bq, aq);
        ak = WMMA_BF16(a, bk, ak);
        av = WMMA_BF16(a, bv, av);
      }
#pragma unroll
      for (int r = 0; r < 8; ++r) {
        int i = mt * 16 + r + half * 8;       // C/D layout: M = r + 8*half
        qh[h * 2560 + i * 40 + nt * 16 + lm]   = f2bf(aq[r] * qscale);
        kh[h * 2560 + i * 40 + nt * 16 + lm]   = f2bf(ak[r]);
        vt[h * 2304 + (nt * 16 + lm) * 72 + i] = f2bf(av[r]);   // transposed
      }
    }
  }
  __syncthreads();

  // ---- Stage 3: sim = q k^T + pos ; softmax, one 16-row strip at a time ----
  // Only 4 sim tiles (32 VGPRs) + one A tile live per strip.
#pragma unroll 1
  for (int mt = 0; mt < 4; ++mt) {
    v16bf aq = load_a(qh + h * 2560 + (mt * 16 + lm) * 40 + half * 16);
    v8f s[4];
#pragma unroll
    for (int jt = 0; jt < 4; ++jt) {
      v16bf b = load_b(kh + h * 2560 + (jt * 16 + lm) * 40 + half * 16);
      v8f z = {0,0,0,0,0,0,0,0};
      s[jt] = WMMA_BF16(aq, b, z);            // K=32 -> single WMMA
    }
#pragma unroll
    for (int jt = 0; jt < 4; ++jt)
#pragma unroll
      for (int r = 0; r < 8; ++r)
        s[jt][r] += pos[(h * 64 + mt * 16 + r + half * 8) * 64 + jt * 16 + lm];
#pragma unroll
    for (int r = 0; r < 8; ++r) {
      // row (mt*16 + r + 8*half) lives entirely in this 16-lane half
      float m = fmaxf(fmaxf(s[0][r], s[1][r]), fmaxf(s[2][r], s[3][r]));
      m = fmaxf(m, __shfl_xor(m, 1, 32));
      m = fmaxf(m, __shfl_xor(m, 2, 32));
      m = fmaxf(m, __shfl_xor(m, 4, 32));
      m = fmaxf(m, __shfl_xor(m, 8, 32));
      float sum = 0.f;
#pragma unroll
      for (int jt = 0; jt < 4; ++jt) {
        float p = __expf(s[jt][r] - m);
        s[jt][r] = p;
        sum += p;
      }
      sum += __shfl_xor(sum, 1, 32);
      sum += __shfl_xor(sum, 2, 32);
      sum += __shfl_xor(sum, 4, 32);
      sum += __shfl_xor(sum, 8, 32);
      float inv = 1.0f / sum;
      int i = mt * 16 + r + half * 8;
#pragma unroll
      for (int jt = 0; jt < 4; ++jt)
        att[h * 4608 + i * 72 + jt * 16 + lm] = f2bf(s[jt][r] * inv);
    }
  }
  __syncthreads();

  // ---- Stage 4: out_h = attn @ v  (K=64 -> two WMMA steps) ----
#pragma unroll 1
  for (int mt = 0; mt < 4; ++mt) {
#pragma unroll
    for (int dt = 0; dt < 2; ++dt) {
      v8f acc = {0,0,0,0,0,0,0,0};
#pragma unroll
      for (int kb = 0; kb < 2; ++kb) {
        v16bf a = load_a(att + h * 4608 + (mt * 16 + lm) * 72 + kb * 32 + half * 16);
        v16bf b = load_b(vt  + h * 2304 + (dt * 16 + lm) * 72 + kb * 32 + half * 16);
        acc = WMMA_BF16(a, b, acc);
      }
#pragma unroll
      for (int r = 0; r < 8; ++r) {
        int i = mt * 16 + r + half * 8;
        oat[i * 264 + h * 32 + dt * 16 + lm] = f2bf(acc[r]);
      }
    }
  }
  __syncthreads();

  // ---- Stage 5: final projection + bias, scatter back to (b,H,W,c) ----
#pragma unroll 1
  for (int t = 0; t < 8; ++t) {
    int tileid = wid * 8 + t;                 // 64 tiles over 8 waves
    int mt = tileid >> 4, nt = tileid & 15;
    v8f acc = {0,0,0,0,0,0,0,0};
#pragma unroll
    for (int kb = 0; kb < 8; ++kb) {
      v16bf a = load_a(oat + (mt * 16 + lm) * 264 + kb * 32 + half * 16);
      v16bf b = load_b(WpT + (nt * 16 + lm) * 256 + kb * 32 + half * 16);
      acc = WMMA_BF16(a, b, acc);
    }
    float bias = bp[nt * 16 + lm];
#pragma unroll
    for (int r = 0; r < 8; ++r) {
      int i = mt * 16 + r + half * 8;
      int grow = nh * 8 + (i >> 3);
      int gcol = nw * 8 + (i & 7);
      out[((((long)bi * 256 + grow) * 256) + gcol) * 256 + nt * 16 + lm] =
          acc[r] + bias;
    }
  }
}

extern "C" void kernel_launch(void* const* d_in, const int* in_sizes, int n_in,
                              void* d_out, int out_size, void* d_ws, size_t ws_size,
                              hipStream_t stream) {
  const float* x   = (const float*)d_in[0];
  const float* Wq  = (const float*)d_in[1];
  const float* Wkv = (const float*)d_in[2];
  const float* pos = (const float*)d_in[3];
  const float* Wp  = (const float*)d_in[4];
  const float* bp  = (const float*)d_in[5];
  float* out = (float*)d_out;

  unsigned short* wsT = (unsigned short*)d_ws;     // 512 KB of bf16 weights
  cvt_weights<<<1024, 256, 0, stream>>>(Wq, Wkv, Wp, wsT);

  const unsigned short* WqT  = wsT;
  const unsigned short* WkvT = wsT + 65536;
  const unsigned short* WpT  = wsT + 196608;

  const size_t lds_bytes = (size_t)130048 * sizeof(unsigned short);  // ~254 KB
  hipFuncSetAttribute((const void*)win_attn,
                      hipFuncAttributeMaxDynamicSharedMemorySize,
                      (int)lds_bytes);
  win_attn<<<4096, 256, lds_bytes, stream>>>(x, pos, bp, WqT, WkvT, WpT, out);
}
